// ComplexGraph_17772574671130
// MI455X (gfx1250) — compile-verified, matrix operand order checked by reading, not run on hardware
//
#include <hip/hip_runtime.h>
#include <hip/hip_bf16.h>
#include <math.h>

typedef __attribute__((ext_vector_type(2)))  float     v2f;
typedef __attribute__((ext_vector_type(8)))  float     v8f;
typedef __attribute__((ext_vector_type(16))) _Float16  v16h;

#define BIGF 1e10f
#define NCOL_MAX 3072      // max residues per row-block LDS row (N=3000 in harness)
#define KMAX 64
#define BLOCK 512          // 16 wave32s per workgroup

#if defined(__has_builtin)
#if __has_builtin(__builtin_amdgcn_wmma_f32_16x16x4_f32)
#define HAVE_WMMA_F32X4 1
#endif
#if __has_builtin(__builtin_amdgcn_update_dpp)
#define HAVE_DPP 1
#endif
#endif

// ---------------------------------------------------------------------------
// Kernel 1: node features H[N, 2E]
// ---------------------------------------------------------------------------
__global__ void cg_feat_kernel(const int* __restrict__ S, const int* __restrict__ RP,
                               const int* __restrict__ ID,
                               const float* __restrict__ res_embed,
                               const float* __restrict__ id_embed,
                               float* __restrict__ out, int N, int E) {
  int i = blockIdx.x;
  int t = threadIdx.x;              // blockDim.x == 2E
  if (i >= N) return;
  if (t < E) {
    out[i * 2 * E + t] = res_embed[S[i] * E + t];
  } else {
    int u = t - E;
    int p = u >> 1;
    float freq = expf((-2.0f * (float)p / (float)E) * 9.210340371976184f);
    float ang  = (float)RP[i] * freq;
    float v    = (u & 1) ? cosf(ang) : sinf(ang);
    out[i * 2 * E + t] = v + id_embed[ID[i] * E + u];
  }
}

// ---------------------------------------------------------------------------
// Dihedral helpers
// ---------------------------------------------------------------------------
struct F3 { float x, y, z; };
__device__ __forceinline__ F3 f3sub(F3 a, F3 b) { return {a.x-b.x, a.y-b.y, a.z-b.z}; }
__device__ __forceinline__ F3 f3cross(F3 a, F3 b) {
  return {a.y*b.z - a.z*b.y, a.z*b.x - a.x*b.z, a.x*b.y - a.y*b.x};
}
__device__ __forceinline__ float f3dot(F3 a, F3 b) { return a.x*b.x + a.y*b.y + a.z*b.z; }
__device__ __forceinline__ F3 atomOf(const float* __restrict__ X, int res, int ch) {
  int r = (res * 4 + ch) * 3;
  return {X[r], X[r+1], X[r+2]};
}
__device__ __forceinline__ float dihedral(F3 a, F3 b, F3 c, F3 d) {
  F3 u1 = f3sub(b, a), u2 = f3sub(c, b), u3 = f3sub(d, c);
  F3 n1 = f3cross(u1, u2), n2 = f3cross(u2, u3);
  float inv = 1.0f / (sqrtf(f3dot(u2, u2)) + 1e-12f);
  F3 u2n = {u2.x*inv, u2.y*inv, u2.z*inv};
  F3 m = f3cross(n1, u2n);
  return atan2f(f3dot(m, n2), f3dot(n1, n2));
}

// min over groups of 4 lanes (column-channel reduction) — DPP quad_perm, no DS waits
__device__ __forceinline__ float quad_min(float v) {
#if HAVE_DPP
  int t = __builtin_amdgcn_update_dpp(0, __float_as_int(v), 0xB1, 0xF, 0xF, true); // {1,0,3,2}
  v = fminf(v, __int_as_float(t));
  t = __builtin_amdgcn_update_dpp(0, __float_as_int(v), 0x4E, 0xF, 0xF, true);     // {2,3,0,1}
  v = fminf(v, __int_as_float(t));
#else
  v = fminf(v, __shfl_xor(v, 1, 32));
  v = fminf(v, __shfl_xor(v, 2, 32));
#endif
  return v;
}

// ---------------------------------------------------------------------------
// Kernel 2: fused WMMA pairwise-min distance + kNN + dihedrals.
// One workgroup (16 waves) per 4-residue row block (16 rows of Xf=[4N,4]).
// Gram tiles via V_WMMA_F32_16X16X4_F32 (xyz padded to K=4, exact fp32).
// Top-k operates on squared distance (monotonic in d, sqrt elided).
// ---------------------------------------------------------------------------
__global__ __launch_bounds__(BLOCK) void cg_knn_kernel(
    const float* __restrict__ X, const int* __restrict__ Seg,
    const int* __restrict__ bid, const int* __restrict__ kptr,
    float* __restrict__ out, int N, int E) {

  __shared__ float s_dist[4 * NCOL_MAX];   // 4 squared-distance rows
  __shared__ float s_sqA[16];
  __shared__ float s_rv[BLOCK / 32];
  __shared__ int   s_ri[BLOCK / 32];
  __shared__ int   s_sel[4 * KMAX];
  __shared__ float s_selv[4 * KMAX];

  const int tid    = threadIdx.x;
  const int lane   = tid & 31;
  const int wave   = tid >> 5;
  const int nwaves = BLOCK >> 5;
  const int B      = blockIdx.x;     // 4-residue block index
  const int i0     = B * 4;
  const int rowBase = B * 16;        // first Xf row of the A tile
  const int K      = *kptr;
  const int rows4  = 4 * N;          // Xf row count
  const int nTiles = (rows4 + 15) >> 4;

  // --- per-lane A-row data (row m = rowBase + (lane%16)); clamp is safe:
  // clamped rows belong to residues >= N whose outputs are masked later.
  const int q  = lane & 15;
  const bool lo = lane < 16;
  int am = rowBase + q; if (am > rows4 - 1) am = rows4 - 1;
  float ax = X[am*3], ay = X[am*3+1], az = X[am*3+2];

  if (tid < 16) {
    int m = rowBase + tid; if (m > rows4 - 1) m = rows4 - 1;
    float x = X[m*3], y = X[m*3+1], z = X[m*3+2];
    s_sqA[tid] = x*x + y*y + z*z;
  }

  // row residue bid/Seg for rule mask (lanes<16 handle rows 0,1; >=16 rows 2,3)
  int r0 = lo ? 0 : 2;
  int ia = i0 + r0, ib = i0 + r0 + 1;
  int bidA = (ia < N) ? bid[ia] : -1, segA = (ia < N) ? Seg[ia] : -1;
  int bidB = (ib < N) ? bid[ib] : -1, segB = (ib < N) ? Seg[ib] : -1;

  __syncthreads();

  float sqa[8];
#pragma unroll
  for (int r = 0; r < 8; ++r) sqa[r] = s_sqA[lo ? r : (r + 8)];

#if HAVE_WMMA_F32X4
  // f32 A-frag 16x4: lanes0-15 hold K0,K1; lanes16-31 hold K2,K3 (same M rows)
  v2f afrag = { lo ? ax : az, lo ? ay : 0.f };
#else
  v16h afrag = {};
  afrag[0] = lo ? (_Float16)ax : (_Float16)0.f;
  afrag[1] = lo ? (_Float16)ay : (_Float16)0.f;
  afrag[2] = lo ? (_Float16)az : (_Float16)0.f;
#endif
  v8f czero = {};

  for (int J = wave; J < nTiles; J += nwaves) {
    int cm = J * 16 + q; if (cm > rows4 - 1) cm = rows4 - 1;
    // prefetch next column tile for this wave
    int pm = cm + nwaves * 16; if (pm > rows4 - 1) pm = rows4 - 1;
    __builtin_prefetch(&X[pm * 3], 0, 1);

    float bx = X[cm*3], by = X[cm*3+1], bz = X[cm*3+2];
    float sqb = bx*bx + by*by + bz*bz;

#if HAVE_WMMA_F32X4
    v2f bfrag = { lo ? bx : bz, lo ? by : 0.f };
    v8f c = __builtin_amdgcn_wmma_f32_16x16x4_f32(
        false, afrag, false, bfrag, (short)0, czero, false, false);
#else
    v16h bfrag = {};
    bfrag[0] = lo ? (_Float16)bx : (_Float16)0.f;
    bfrag[1] = lo ? (_Float16)by : (_Float16)0.f;
    bfrag[2] = lo ? (_Float16)bz : (_Float16)0.f;
    v8f c = __builtin_amdgcn_wmma_f32_16x16x32_f16(
        false, afrag, false, bfrag, (short)0, czero, false, false);
#endif

    // d2[m][n] = sqA[m] + sqB[n] - 2*dot; min over row-residue channels (VGPRs)
    float dlo = sqa[0] - 2.f*c[0];
    dlo = fminf(dlo, sqa[1] - 2.f*c[1]);
    dlo = fminf(dlo, sqa[2] - 2.f*c[2]);
    dlo = fminf(dlo, sqa[3] - 2.f*c[3]);
    float dhi = sqa[4] - 2.f*c[4];
    dhi = fminf(dhi, sqa[5] - 2.f*c[5]);
    dhi = fminf(dhi, sqa[6] - 2.f*c[6]);
    dhi = fminf(dhi, sqa[7] - 2.f*c[7]);
    dlo += sqb; dhi += sqb;

    // min over column-residue channels (groups of 4 lanes) — DPP, no dscnt waits
    dlo = quad_min(dlo);
    dhi = quad_min(dhi);

    if ((lane & 3) == 0) {
      int jres = J * 4 + (q >> 2);
      if (jres < N && jres < NCOL_MAX) {
        int bj = bid[jres], sj = Seg[jres];
        float d0 = (bj == bidA && sj == segA) ? fmaxf(dlo, 0.f) : BIGF;
        float d1 = (bj == bidB && sj == segB) ? fmaxf(dhi, 0.f) : BIGF;
        s_dist[r0 * NCOL_MAX + jres]       = d0;
        s_dist[(r0 + 1) * NCOL_MAX + jres] = d1;
      }
    }
  }
  __syncthreads();

  // --- cooperative top-k (smallest-k, ties -> lowest index) per row ---
  const int ncols = (N < NCOL_MAX) ? N : NCOL_MAX;
  for (int row = 0; row < 4; ++row) {
    float* drow = &s_dist[row * NCOL_MAX];
    for (int t = 0; t < K && t < KMAX; ++t) {
      float bv = 3.0e38f; int bi = 0x7fffffff;
      for (int j = tid; j < ncols; j += BLOCK) {
        float v = drow[j];
        if (v < bv || (v == bv && j < bi)) { bv = v; bi = j; }
      }
      // wave-level min-reduce (no barriers)
#pragma unroll
      for (int off = 16; off > 0; off >>= 1) {
        float ov = __shfl_xor(bv, off, 32);
        int   oi = __shfl_xor(bi, off, 32);
        if (ov < bv || (ov == bv && oi < bi)) { bv = ov; bi = oi; }
      }
      if (lane == 0) { s_rv[wave] = bv; s_ri[wave] = bi; }
      __syncthreads();
      if (tid == 0) {
        float wv = s_rv[0]; int wi = s_ri[0];
        for (int w = 1; w < nwaves; ++w) {
          float v2 = s_rv[w]; int i2 = s_ri[w];
          if (v2 < wv || (v2 == wv && i2 < wi)) { wv = v2; wi = i2; }
        }
        s_sel[row * KMAX + t]  = wi;
        s_selv[row * KMAX + t] = wv;
        if (wi >= 0 && wi < ncols) drow[wi] = 2.0f * BIGF;   // exclude
      }
      __syncthreads();
    }
  }

  // --- edge outputs + dihedrals ---
  const long HOFF = (long)N * 2 * E;
  const long NK   = (long)N * K;
  float* eDst   = out + HOFF;
  float* eSrc   = out + HOFF + NK;
  float* eAttr  = out + HOFF + 2 * NK;
  float* eValid = out + HOFF + 4 * NK;

  for (int t = tid; t < 4 * K; t += BLOCK) {
    int row = t / K, slot = t - row * K;
    int ires = i0 + row;
    if (ires >= N || slot >= KMAX) continue;
    long e   = (long)ires * K + slot;
    float d2 = s_selv[row * KMAX + slot];
    int  idx = s_sel[row * KMAX + slot];
    bool ok  = (d2 < BIGF);            // squared-distance sentinel, same validity
    int dst = ok ? idx : 0;
    int src = ires;
    float phi = 0.f, psi = 0.f;
    if (ok) {
      phi = dihedral(atomOf(X, dst, 2), atomOf(X, src, 0),
                     atomOf(X, src, 1), atomOf(X, src, 2));
      psi = dihedral(atomOf(X, dst, 0), atomOf(X, dst, 1),
                     atomOf(X, dst, 2), atomOf(X, src, 0));
    }
    eDst[e]        = ok ? (float)idx  : -1.f;
    eSrc[e]        = ok ? (float)ires : -1.f;
    eAttr[2*e]     = phi;
    eAttr[2*e + 1] = psi;
    eValid[e]      = ok ? 1.f : 0.f;
  }
}

// ---------------------------------------------------------------------------
extern "C" void kernel_launch(void* const* d_in, const int* in_sizes, int n_in,
                              void* d_out, int out_size, void* d_ws, size_t ws_size,
                              hipStream_t stream) {
  (void)n_in; (void)out_size; (void)d_ws; (void)ws_size;
  const float* X         = (const float*)d_in[0];
  const int*   S         = (const int*)d_in[1];
  const int*   RP        = (const int*)d_in[2];
  const int*   ID        = (const int*)d_in[3];
  const int*   Seg       = (const int*)d_in[4];
  const int*   bid       = (const int*)d_in[5];
  const float* res_embed = (const float*)d_in[6];
  const float* id_embed  = (const float*)d_in[7];
  const int*   kptr      = (const int*)d_in[8];

  const int N = in_sizes[1];          // residues
  const int E = in_sizes[7] / 2;      // id_embed is [NUM_ID=2, E]
  float* out = (float*)d_out;

  cg_feat_kernel<<<N, 2 * E, 0, stream>>>(S, RP, ID, res_embed, id_embed, out, N, E);

  const int nBlocks = (N + 3) / 4;
  cg_knn_kernel<<<nBlocks, BLOCK, 0, stream>>>(X, Seg, bid, kptr, out, N, E);
}
